// LSTMModel_4518305595962
// MI455X (gfx1250) — matrix operand order repaired and checked by reading, not compiled, not run on hardware
//
#include <hip/hip_runtime.h>

#define HID 64

typedef _Float16 v8h  __attribute__((ext_vector_type(8)));
typedef _Float16 v16h __attribute__((ext_vector_type(16)));
typedef float    v2f  __attribute__((ext_vector_type(2)));
typedef float    v8f  __attribute__((ext_vector_type(8)));

union V16U { v16h v; v8h h[2]; };

#if __has_builtin(__builtin_amdgcn_wmma_f32_16x16x4_f32)
#define HAVE_WMMA_F32X4 1
#else
#define HAVE_WMMA_F32X4 0
#endif

__device__ __forceinline__ float fast_rcp(float x) {
#if __has_builtin(__builtin_amdgcn_rcpf)
  return __builtin_amdgcn_rcpf(x);     // v_rcp_f32
#else
  return 1.0f / x;
#endif
}
__device__ __forceinline__ float fast_tanh(float x) {
#if __has_builtin(__builtin_amdgcn_tanhf)
  return __builtin_amdgcn_tanhf(x);    // gfx1250 v_tanh_f32 (1 TRANS op)
#else
  return 2.0f * fast_rcp(1.0f + __expf(-2.0f * x)) - 1.0f;
#endif
}
__device__ __forceinline__ float fast_sigmoid(float x) {
#if __has_builtin(__builtin_amdgcn_tanhf)
  return 0.5f + 0.5f * __builtin_amdgcn_tanhf(0.5f * x);   // mul+tanh+fma
#else
  return fast_rcp(1.0f + __expf(-x));
#endif
}
__device__ __forceinline__ v8f wmma_f16(v16h a, v16h b, v8f c) {
  // v_wmma_f32_16x16x32_f16  (neg_a, A, neg_b, B, c_mod, C, reuse_a, reuse_b)
  return __builtin_amdgcn_wmma_f32_16x16x32_f16(false, a, false, b, (short)0, c,
                                                false, false);
}

// ---------------------------------------------------------------------------
// Prep: f32 -> f16 weights; fold decoder Wih+Whh (input==h); fold biases;
// zero-pad lin_W to a [16 x 64] B-matrix for the output-head WMMA.
// ---------------------------------------------------------------------------
__global__ void lstm_prep(const float* __restrict__ encWhh,
                          const float* __restrict__ decWih,
                          const float* __restrict__ decWhh,
                          const float* __restrict__ encBih,
                          const float* __restrict__ encBhh,
                          const float* __restrict__ decBih,
                          const float* __restrict__ decBhh,
                          const float* __restrict__ linW,
                          _Float16* __restrict__ wEnc,
                          _Float16* __restrict__ wDec,
                          _Float16* __restrict__ linPad,
                          float* __restrict__ biasE,
                          float* __restrict__ biasD) {
  const int idx = blockIdx.x * blockDim.x + threadIdx.x;
  if (idx < 256 * HID) {
    wEnc[idx] = (_Float16)encWhh[idx];
    wDec[idx] = (_Float16)(decWih[idx] + decWhh[idx]);
  }
  if (idx < 16 * HID) {
    const int n = idx / HID, k = idx % HID;
    linPad[idx] = (n < 2) ? (_Float16)linW[n * HID + k] : (_Float16)0.0f;
  }
  if (idx < 256) {
    biasE[idx] = encBih[idx] + encBhh[idx];
    biasD[idx] = decBih[idx] + decBhh[idx];
  }
}

// ---------------------------------------------------------------------------
// Persistent LSTM: one wave <-> 16 samples, h/c resident across all 50 steps.
// gates[16x256] = h[16x64] @ W^T  via v_wmma_f32_16x16x32_f16 (f32 accumulate).
// Bias (and the encoder's x @ Wih^T) are produced by v_wmma_f32_16x16x4_f32
// with C = inline 0: A column K2 is forced to 1.0 and B row 2 carries bias[g],
// so no per-tile VGPR broadcast movs are needed.
// ---------------------------------------------------------------------------
__launch_bounds__(256)
__global__ void lstm_persist(const float* __restrict__ x_in,       // [B,20,2]
                             const float* __restrict__ encWih,     // [256,2]
                             const _Float16* __restrict__ wEnc,    // [256,64]
                             const _Float16* __restrict__ wDec,    // [256,64]
                             const _Float16* __restrict__ linPad,  // [16,64]
                             const float* __restrict__ biasE,      // [256]
                             const float* __restrict__ biasD,      // [256]
                             const float* __restrict__ linb,       // [2]
                             float* __restrict__ out) {            // [B,30,2]
  __shared__ __align__(32) _Float16 sW[256 * HID];    // 32 KB (enc then dec)
  __shared__ __align__(32) _Float16 sLin[16 * HID];   // 2 KB
  __shared__ float sBias[512];                        // 2 KB ([256..511] = 0)
  __shared__ float sWih[256 * 2];                     // 2 KB
  __shared__ __align__(32) _Float16 sH[8][16 * HID];  // 16 KB, per-wave h (f16)

  const int tid  = threadIdx.x;
  const int lane = tid & 31;
  const int wave = tid >> 5;
  const int row  = lane & 15;   // M for A-frags, N for B/C/D frags
  const int hi   = lane >> 4;   // lane half (K split per ISA layouts)
  const long sampleBase = (long)blockIdx.x * 128 + (long)wave * 16;

  // ---- stage encoder constants ----
  {
    const uint4* src = (const uint4*)wEnc;   // 2048 x 16B
    uint4* dst = (uint4*)sW;
    for (int i = tid; i < 2048; i += 256) dst[i] = src[i];
    const uint4* ls = (const uint4*)linPad;  // 128 x 16B
    uint4* ld = (uint4*)sLin;
    for (int i = tid; i < 128; i += 256) ld[i] = ls[i];
    sBias[tid]       = biasE[tid];
    sBias[tid + 256] = 0.0f;                 // zero pad for hi-lane selects
    sWih[tid]       = encWih[tid];
    sWih[tid + 256] = encWih[tid + 256];
    float* hz = (float*)&sH[wave][0];        // zero own h staging (512 f32)
    for (int i = lane; i < 512; i += 32) hz[i] = 0.0f;
  }
  __syncthreads();

  // A-fragment of h (16x32 f16): lane holds its row; K-halves split by lane half.
  auto loadH = [&](int kc) -> v16h {
    const _Float16* hr = &sH[wave][row * HID];
    V16U a;
    a.h[0] = *(const v8h*)(hr + kc + hi * 16);       // VGPR0-3: K=kc+0..7 (or 16..23)
    a.h[1] = *(const v8h*)(hr + kc + hi * 16 + 8);   // VGPR4-7: K=kc+8..15 (or 24..31)
    return a.v;
  };
  // B-fragment (32x16 f16): lane holds column (n*16+row); 16 contiguous K values.
  auto loadB = [&](const _Float16* base, int n, int kc) -> v16h {
    return *(const v16h*)(base + (n * 16 + row) * HID + kc + hi * 16);
  };

  float c_reg[4][8];
#pragma unroll
  for (int t = 0; t < 4; ++t)
#pragma unroll
    for (int r = 0; r < 8; ++r) c_reg[t][r] = 0.0f;

  const float oneHi = hi ? 1.0f : 0.0f;   // A column K2 = 1.0 (bias path)

  // =========================== encoder: 20 steps ===========================
#pragma clang loop unroll(disable)
  for (int s = 0; s < 20; ++s) {
    // Block LICM: keep weight fragments in LDS, not hoisted into 256+ VGPRs.
    asm volatile("" ::: "memory");
#if HAVE_WMMA_F32X4
    // A 16x4 f32: lanes<16 hold K0,K1 = x ; lanes>=16 hold K2=1.0, K3=0.
    v2f ax = {oneHi, 0.0f};
    if (hi == 0) {
      const float2 xv = *(const float2*)(x_in + (sampleBase + row) * 40 + s * 2);
      ax.x = xv.x; ax.y = xv.y;
    }
#else
    float2 xr[8];
#pragma unroll
    for (int r = 0; r < 8; ++r)
      xr[r] = *(const float2*)(x_in + (sampleBase + hi * 8 + r) * 40 + s * 2);
#endif
    const v16h hA0 = loadH(0), hA1 = loadH(32);
#pragma unroll
    for (int t = 0; t < 4; ++t) {           // hidden tile (16 cols of h)
      v8f acc[4];
#pragma unroll
      for (int q = 0; q < 4; ++q) {         // i, f, g, o quadrant
        const int n = q * 4 + t;            // gate tile
        const int g = n * 16 + row;
        v8f a = {};                          // C = inline 0
#if HAVE_WMMA_F32X4
        // B rows: 0,1 = Wih columns; row 2 = bias (hit by A's K2 = 1.0).
        v2f bx = { sWih[g * 2 + hi], sBias[hi * 256 + g] };
        a = __builtin_amdgcn_wmma_f32_16x16x4_f32(false, ax, false, bx,
                                                  (short)0, a, false, false);
#else
        {
          const float b  = sBias[g];
          const float w0 = sWih[g * 2 + 0];
          const float w1 = sWih[g * 2 + 1];
#pragma unroll
          for (int r = 0; r < 8; ++r) a[r] = b + w0 * xr[r].x + w1 * xr[r].y;
        }
#endif
        a = wmma_f16(hA0, loadB(sW, n, 0), a);
        a = wmma_f16(hA1, loadB(sW, n, 32), a);
        acc[q] = a;
      }
#pragma unroll
      for (int r = 0; r < 8; ++r) {
        const float iv = fast_sigmoid(acc[0][r]);
        const float fv = fast_sigmoid(acc[1][r]);
        const float gv = fast_tanh(acc[2][r]);
        const float ov = fast_sigmoid(acc[3][r]);
        const float cn = fv * c_reg[t][r] + iv * gv;
        c_reg[t][r] = cn;
        sH[wave][(hi * 8 + r) * HID + t * 16 + row] =
            (_Float16)(ov * fast_tanh(cn));
      }
    }
  }

  // ---- swap to decoder weights (reuse sW/sBias to stay under LDS limit) ----
  __syncthreads();
  {
    const uint4* src = (const uint4*)wDec;
    uint4* dst = (uint4*)sW;
    for (int i = tid; i < 2048; i += 256) dst[i] = src[i];
    sBias[tid] = biasD[tid];                // upper half stays zero
  }
  __syncthreads();

  // Bias / lin_b injection operands for the decoder (constant per thread).
  const v2f axb = {oneHi, 0.0f};                                   // K2 = 1.0
  const float lbv = (hi == 0 && row < 2) ? linb[row] : 0.0f;
  const v2f bxo = {0.0f, lbv};                                     // lin_b row

  // =========================== decoder: 30 steps ===========================
#pragma clang loop unroll(disable)
  for (int s = 0; s < 30; ++s) {
    asm volatile("" ::: "memory");          // block LICM of LDS weight reads
    const v16h hA0 = loadH(0), hA1 = loadH(32);
#pragma unroll
    for (int t = 0; t < 4; ++t) {
      v8f acc[4];
#pragma unroll
      for (int q = 0; q < 4; ++q) {
        const int n = q * 4 + t;
        const int g = n * 16 + row;
        v8f a = {};                          // C = inline 0
#if HAVE_WMMA_F32X4
        v2f bxd = {0.0f, sBias[hi * 256 + g]};   // bias-only injection
        a = __builtin_amdgcn_wmma_f32_16x16x4_f32(false, axb, false, bxd,
                                                  (short)0, a, false, false);
#else
        {
          const float b = sBias[g];
#pragma unroll
          for (int r = 0; r < 8; ++r) a[r] = b;
        }
#endif
        a = wmma_f16(hA0, loadB(sW, n, 0), a);   // combined (Wih+Whh), x==h
        a = wmma_f16(hA1, loadB(sW, n, 32), a);
        acc[q] = a;
      }
#pragma unroll
      for (int r = 0; r < 8; ++r) {
        const float iv = fast_sigmoid(acc[0][r]);
        const float fv = fast_sigmoid(acc[1][r]);
        const float gv = fast_tanh(acc[2][r]);
        const float ov = fast_sigmoid(acc[3][r]);
        const float cn = fv * c_reg[t][r] + iv * gv;
        c_reg[t][r] = cn;
        sH[wave][(hi * 8 + r) * HID + t * 16 + row] =
            (_Float16)(ov * fast_tanh(cn));
      }
    }
    // output head: out[16x2] = h_new @ lin_W^T + lin_b, via padded 16x64 WMMA
    const v16h nh0 = loadH(0), nh1 = loadH(32);
    v8f oa = {};
#if HAVE_WMMA_F32X4
    oa = __builtin_amdgcn_wmma_f32_16x16x4_f32(false, axb, false, bxo,
                                               (short)0, oa, false, false);
#else
    {
      const float b = (row < 2) ? linb[row] : 0.0f;
#pragma unroll
      for (int r = 0; r < 8; ++r) oa[r] = b;
    }
#endif
    oa = wmma_f16(nh0, loadB(sLin, 0, 0), oa);
    oa = wmma_f16(nh1, loadB(sLin, 0, 32), oa);
    if (row < 2) {  // D-frag columns N=0,1 hold the two outputs
#pragma unroll
      for (int r = 0; r < 8; ++r) {
        const long m = sampleBase + hi * 8 + r;
        out[m * 60 + s * 2 + row] = oa[r];
      }
    }
  }
}

// ---------------------------------------------------------------------------
extern "C" void kernel_launch(void* const* d_in, const int* in_sizes, int n_in,
                              void* d_out, int out_size, void* d_ws, size_t ws_size,
                              hipStream_t stream) {
  const float* train  = (const float*)d_in[0];
  const float* encWih = (const float*)d_in[1];
  const float* encWhh = (const float*)d_in[2];
  const float* encBih = (const float*)d_in[3];
  const float* encBhh = (const float*)d_in[4];
  const float* decWih = (const float*)d_in[5];
  const float* decWhh = (const float*)d_in[6];
  const float* decBih = (const float*)d_in[7];
  const float* decBhh = (const float*)d_in[8];
  const float* linW   = (const float*)d_in[9];
  const float* linb   = (const float*)d_in[10];
  float* out = (float*)d_out;

  const int B = in_sizes[0] / 40;  // [B,20,2]

  char* ws = (char*)d_ws;
  _Float16* wEnc   = (_Float16*)(ws);           // 32768 B
  _Float16* wDec   = (_Float16*)(ws + 32768);   // 32768 B
  _Float16* linPad = (_Float16*)(ws + 65536);   //  2048 B
  float*    biasE  = (float*)   (ws + 67584);   //  1024 B
  float*    biasD  = (float*)   (ws + 68608);   //  1024 B

  lstm_prep<<<64, 256, 0, stream>>>(encWhh, decWih, decWhh, encBih, encBhh,
                                    decBih, decBhh, linW, wEnc, wDec, linPad,
                                    biasE, biasD);
  lstm_persist<<<B / 128, 256, 0, stream>>>(train, encWih, wEnc, wDec, linPad,
                                            biasE, biasD, linb, out);
}
